// FCOSDetect_47339129536829
// MI455X (gfx1250) — 8-bit matrix operand order repaired and checked by reading, compiled, not run on hardware
//
#include <hip/hip_runtime.h>
#include <hip/hip_bf16.h>
#include <float.h>

#define B_IMG       16
#define N_LOC       17064
#define NUM_CLASSES 80
#define MAX_NUM     1000
#define NPAD        1024
#define SCORE_THR   0.05f
#define IOU_THR     0.6f
#define CAP         2048
#define NBINS       2048

typedef __attribute__((ext_vector_type(8))) int v8i;

__device__ __forceinline__ float sigmoidf(float x) { return 1.0f / (1.0f + expf(-x)); }

__global__ void fcos_score_kernel(
    const float* __restrict__ cls0, const float* __restrict__ cls1,
    const float* __restrict__ cls2, const float* __restrict__ cls3,
    const float* __restrict__ cls4,
    const float* __restrict__ reg0, const float* __restrict__ reg1,
    const float* __restrict__ reg2, const float* __restrict__ reg3,
    const float* __restrict__ reg4,
    const float* __restrict__ ctr0, const float* __restrict__ ctr1,
    const float* __restrict__ ctr2, const float* __restrict__ ctr3,
    const float* __restrict__ ctr4,
    float* __restrict__ s_ws, int* __restrict__ l_ws, float* __restrict__ b_ws)
{
  int gid = blockIdx.x * blockDim.x + threadIdx.x;
  if (gid >= B_IMG * N_LOC) return;
  int b   = gid / N_LOC;
  int loc = gid - b * N_LOC;

  const float* clsTab[5] = {cls0, cls1, cls2, cls3, cls4};
  const float* regTab[5] = {reg0, reg1, reg2, reg3, reg4};
  const float* ctrTab[5] = {ctr0, ctr1, ctr2, ctr3, ctr4};
  const int offs[6]    = {0, 12800, 16000, 16800, 17008, 17064};
  const int widths[5]  = {128, 64, 32, 16, 8};
  const int strides[5] = {8, 16, 32, 64, 128};
  const int hwTab[5]   = {12800, 3200, 800, 208, 56};

  int lvl = 0;
  #pragma unroll
  for (int l = 1; l < 5; ++l) if (loc >= offs[l]) lvl = l;
  int pos = loc - offs[lvl];
  int hw  = hwTab[lvl];
  int w   = widths[lvl];
  int s   = strides[lvl];
  int iy  = pos / w;
  int ix  = pos - iy * w;
  float cx = (float)(ix * s + (s >> 1));
  float cy = (float)(iy * s + (s >> 1));

  const float* cp = clsTab[lvl] + (size_t)b * NUM_CLASSES * hw + pos;
  float best = cp[0];
  int   bi   = 0;
  for (int c = 1; c < NUM_CLASSES; ++c) {
    if (c + 8 < NUM_CLASSES) __builtin_prefetch(cp + (size_t)(c + 8) * hw, 0, 0);
    float v = cp[(size_t)c * hw];
    if (v > best) { best = v; bi = c; }
  }
  float ctrLogit = ctrTab[lvl][(size_t)b * hw + pos];
  float score = sqrtf(sigmoidf(best) * sigmoidf(ctrLogit));

  const float* rp = regTab[lvl] + (size_t)b * 4 * hw + pos;
  float r0 = rp[0];
  float r1 = rp[(size_t)hw];
  float r2 = rp[(size_t)2 * hw];
  float r3 = rp[(size_t)3 * hw];

  size_t o = (size_t)b * N_LOC + loc;
  s_ws[o]       = score;
  l_ws[o]       = bi + 1;
  b_ws[o*4 + 0] = cx - r0;
  b_ws[o*4 + 1] = cy - r1;
  b_ws[o*4 + 2] = cx + r2;
  b_ws[o*4 + 3] = cy + r3;
}

__global__ void fcos_topk_kernel(
    const float* __restrict__ s_ws, const int* __restrict__ l_ws,
    const float* __restrict__ b_ws,
    float* __restrict__ tk_s, int* __restrict__ tk_l, float* __restrict__ tk_b)
{
  __shared__ int   hist[NBINS];
  __shared__ float cs[CAP];
  __shared__ int   ci[CAP];
  __shared__ int   sTbin;
  __shared__ int   sCnt;

  const int b   = blockIdx.x;
  const int tid = threadIdx.x;
  const int bd  = blockDim.x;
  const float* sc_base = s_ws + (size_t)b * N_LOC;

  for (int i = tid; i < NBINS; i += bd) hist[i] = 0;
  if (tid == 0) sCnt = 0;
  __syncthreads();

  for (int i = tid; i < N_LOC; i += bd) {
    float sc = sc_base[i];
    int bin = (int)(sc * (float)NBINS);
    bin = bin < 0 ? 0 : (bin > NBINS - 1 ? NBINS - 1 : bin);
    atomicAdd(&hist[bin], 1);
  }
  __syncthreads();

  if (tid == 0) {
    int cum = 0, tb = 0;
    for (int bn = NBINS - 1; bn >= 0; --bn) {
      cum += hist[bn];
      if (cum >= MAX_NUM) { tb = bn; break; }
    }
    sTbin = tb;
  }
  __syncthreads();

  int tbin = sTbin;
  for (int i = tid; i < N_LOC; i += bd) {
    float sc = sc_base[i];
    int bin = (int)(sc * (float)NBINS);
    bin = bin < 0 ? 0 : (bin > NBINS - 1 ? NBINS - 1 : bin);
    if (bin >= tbin) {
      int p = atomicAdd(&sCnt, 1);
      if (p < CAP) { cs[p] = sc; ci[p] = i; }
    }
  }
  __syncthreads();

  int total = sCnt < CAP ? sCnt : CAP;
  for (int i = tid; i < CAP; i += bd) {
    if (i >= total) { cs[i] = -1e30f; ci[i] = 0x7fffffff; }
  }
  __syncthreads();

  for (int k = 2; k <= CAP; k <<= 1) {
    for (int j = k >> 1; j > 0; j >>= 1) {
      for (int i = tid; i < CAP; i += bd) {
        int ixj = i ^ j;
        if (ixj > i) {
          float sa = cs[i], sb = cs[ixj];
          int   ia = ci[i], ib = ci[ixj];
          bool up      = ((i & k) == 0);
          bool aBefore = (sa > sb) || (sa == sb && ia < ib);
          bool doSwap  = up ? !aBefore : aBefore;
          if (doSwap) { cs[i] = sb; cs[ixj] = sa; ci[i] = ib; ci[ixj] = ia; }
        }
      }
      __syncthreads();
    }
  }

  for (int r = tid; r < MAX_NUM; r += bd) {
    float sc = cs[r];
    int loc  = ci[r];
    size_t o = (size_t)b * MAX_NUM + r;
    if (loc >= 0 && loc < N_LOC) {
      size_t src = (size_t)b * N_LOC + loc;
      tk_s[o] = sc;
      tk_l[o] = l_ws[src];
      tk_b[o*4 + 0] = b_ws[src*4 + 0];
      tk_b[o*4 + 1] = b_ws[src*4 + 1];
      tk_b[o*4 + 2] = b_ws[src*4 + 2];
      tk_b[o*4 + 3] = b_ws[src*4 + 3];
    } else {
      tk_s[o] = -1e30f; tk_l[o] = 0;
      tk_b[o*4+0] = 0.f; tk_b[o*4+1] = 0.f; tk_b[o*4+2] = 0.f; tk_b[o*4+3] = 0.f;
    }
  }
}

__global__ void fcos_nms_kernel(
    const float* __restrict__ tk_s, const int* __restrict__ tk_l,
    const float* __restrict__ tk_b, float* __restrict__ out)
{
  __shared__ float sS[NPAD];
  __shared__ int   sL[NPAD];
  __shared__ float sX1[NPAD], sY1[NPAD], sX2[NPAD], sY2[NPAD];
  __shared__ float sArea[NPAD];
  __shared__ unsigned char sKeep[NPAD];
  __shared__ unsigned int maskPack[16 * (NPAD / 4)];
  __shared__ unsigned int keepPack[NPAD / 4];
  __shared__ int   sSup[16];
  __shared__ float sRed[256];
  __shared__ float sCoordMax;

  const int b   = blockIdx.x;
  const int tid = threadIdx.x;
  const int bd  = blockDim.x;

  for (int r = tid; r < NPAD; r += bd) {
    if (r < MAX_NUM) {
      size_t o = (size_t)b * MAX_NUM + r;
      sS[r]  = tk_s[o];
      sL[r]  = tk_l[o];
      sX1[r] = tk_b[o*4 + 0];
      sY1[r] = tk_b[o*4 + 1];
      sX2[r] = tk_b[o*4 + 2];
      sY2[r] = tk_b[o*4 + 3];
    } else {
      sS[r] = -1e30f; sL[r] = 0;
      sX1[r] = 0.f; sY1[r] = 0.f; sX2[r] = 0.f; sY2[r] = 0.f;
    }
  }
  for (int i = tid; i < NPAD / 4; i += bd) keepPack[i] = 0u;
  __syncthreads();

  float lmax = -FLT_MAX;
  for (int r = tid; r < MAX_NUM; r += bd) {
    float v;
    if (sS[r] > SCORE_THR)
      v = fmaxf(fmaxf(sX1[r], sY1[r]), fmaxf(sX2[r], sY2[r]));
    else
      v = 0.0f;
    lmax = fmaxf(lmax, v);
  }
  sRed[tid] = lmax;
  __syncthreads();
  for (int st = 128; st > 0; st >>= 1) {
    if (tid < st) sRed[tid] = fmaxf(sRed[tid], sRed[tid + st]);
    __syncthreads();
  }
  if (tid == 0) sCoordMax = sRed[0];
  __syncthreads();
  float cmax1 = sCoordMax + 1.0f;

  for (int r = tid; r < NPAD; r += bd) {
    float off = (float)sL[r] * cmax1;
    float x1 = sX1[r] + off, y1 = sY1[r] + off;
    float x2 = sX2[r] + off, y2 = sY2[r] + off;
    sX1[r] = x1; sY1[r] = y1; sX2[r] = x2; sY2[r] = y2;
    sArea[r] = fmaxf(x2 - x1, 0.f) * fmaxf(y2 - y1, 0.f);
    sKeep[r] = (r < MAX_NUM && sS[r] > SCORE_THR) ? 1 : 0;
  }
  __syncthreads();

  for (int t = 0; t < NPAD / 16; ++t) {
    const int limit = t * 16;
    const int dlim  = (limit + 3) >> 2;

    for (int w = tid; w < 16 * dlim; w += bd) {
      int r = w / dlim;
      int d = w - r * dlim;
      int j = limit + r;
      float rx1 = sX1[j], ry1 = sY1[j], rx2 = sX2[j], ry2 = sY2[j];
      float ra  = sArea[j];
      unsigned int word = 0;
      #pragma unroll
      for (int q = 0; q < 4; ++q) {
        int i = 4 * d + q;
        float ix1 = fmaxf(rx1, sX1[i]);
        float iy1 = fmaxf(ry1, sY1[i]);
        float ix2 = fminf(rx2, sX2[i]);
        float iy2 = fminf(ry2, sY2[i]);
        float inter = fmaxf(ix2 - ix1, 0.f) * fmaxf(iy2 - iy1, 0.f);
        float uni   = ra + sArea[i] - inter;
        float iou   = inter / fmaxf(uni, 1e-9f);
        unsigned int m = (iou > IOU_THR) ? 1u : 0u;
        word |= m << (8 * q);
      }
      maskPack[r * (NPAD / 4) + d] = word;
    }
    __syncthreads();

    if (tid < 32) {
      int lane = tid;
      int grp  = lane >> 4;
      int rloc = lane & 15;
      const unsigned int* mrow = &maskPack[rloc * (NPAD / 4)];

      v8i acc = {0, 0, 0, 0, 0, 0, 0, 0};
      for (int kb = 0; kb < limit; kb += 64) {
        int kd = kb >> 2;
        v8i A, Bv;
        #pragma unroll
        for (int v = 0; v < 8; ++v) {
          int ad = kd + ((v >> 2) << 3) + ((v & 2) << 1) + (v & 1) + (grp << 1);
          A[v] = (int)mrow[ad];
        }
        #pragma unroll
        for (int v = 0; v < 8; ++v) {
          int bdw = kd + ((v >> 2) << 3) + ((v & 2) << 1) + (v & 1) + (grp << 1);
          Bv[v] = (int)keepPack[bdw];
        }
        acc = __builtin_amdgcn_wmma_i32_16x16x64_iu8(false, A, false, Bv, acc,
                                                     false, false);
      }
      if (lane == 0 || lane == 16) {
        #pragma unroll
        for (int d = 0; d < 8; ++d) sSup[grp * 8 + d] = acc[d];
      }
    }
    __syncthreads();

    if (tid == 0) {
      for (int r = 0; r < 16; ++r) {
        int j = limit + r;
        bool k = (sKeep[j] != 0) && (sSup[r] == 0);
        if (k) {
          for (int rr = 0; rr < r; ++rr) {
            int jj = limit + rr;
            if (sKeep[jj]) {
              float ix1 = fmaxf(sX1[j], sX1[jj]);
              float iy1 = fmaxf(sY1[j], sY1[jj]);
              float ix2 = fminf(sX2[j], sX2[jj]);
              float iy2 = fminf(sY2[j], sY2[jj]);
              float inter = fmaxf(ix2 - ix1, 0.f) * fmaxf(iy2 - iy1, 0.f);
              float uni   = sArea[j] + sArea[jj] - inter;
              float iou   = inter / fmaxf(uni, 1e-9f);
              if (iou > IOU_THR) { k = false; break; }
            }
          }
        }
        sKeep[j] = k ? 1 : 0;
      }
      #pragma unroll
      for (int q = 0; q < 4; ++q) {
        int base = limit + 4 * q;
        unsigned int word = (unsigned int)sKeep[base]
                          | ((unsigned int)sKeep[base + 1] << 8)
                          | ((unsigned int)sKeep[base + 2] << 16)
                          | ((unsigned int)sKeep[base + 3] << 24);
        keepPack[(limit >> 2) + q] = word;
      }
    }
    __syncthreads();
  }

  float* oS = out;
  float* oL = out + B_IMG * MAX_NUM;
  float* oB = out + 2 * B_IMG * MAX_NUM;
  float* oK = out + 6 * B_IMG * MAX_NUM;
  for (int r = tid; r < MAX_NUM; r += bd) {
    size_t o = (size_t)b * MAX_NUM + r;
    bool kp = sKeep[r] != 0;
    oS[o] = kp ? sS[r] : 0.0f;
    oL[o] = kp ? (float)sL[r] : 0.0f;
    oB[o*4 + 0] = tk_b[o*4 + 0];
    oB[o*4 + 1] = tk_b[o*4 + 1];
    oB[o*4 + 2] = tk_b[o*4 + 2];
    oB[o*4 + 3] = tk_b[o*4 + 3];
    oK[o] = kp ? 1.0f : 0.0f;
  }
}

extern "C" void kernel_launch(void* const* d_in, const int* in_sizes, int n_in,
                              void* d_out, int out_size, void* d_ws, size_t ws_size,
                              hipStream_t stream)
{
  (void)in_sizes; (void)n_in; (void)out_size; (void)ws_size;

  const float* cls[5]; const float* reg[5]; const float* ctr[5];
  for (int i = 0; i < 5; ++i) {
    cls[i] = (const float*)d_in[i];
    reg[i] = (const float*)d_in[5 + i];
    ctr[i] = (const float*)d_in[10 + i];
  }

  float* s_ws = (float*)d_ws;
  int*   l_ws = (int*)  (s_ws + (size_t)B_IMG * N_LOC);
  float* b_ws = (float*)(l_ws + (size_t)B_IMG * N_LOC);
  float* tk_s = b_ws + (size_t)B_IMG * N_LOC * 4;
  int*   tk_l = (int*)  (tk_s + (size_t)B_IMG * MAX_NUM);
  float* tk_b = (float*)(tk_l + (size_t)B_IMG * MAX_NUM);
  float* out  = (float*)d_out;

  int total = B_IMG * N_LOC;
  fcos_score_kernel<<<(total + 255) / 256, 256, 0, stream>>>(
      cls[0], cls[1], cls[2], cls[3], cls[4],
      reg[0], reg[1], reg[2], reg[3], reg[4],
      ctr[0], ctr[1], ctr[2], ctr[3], ctr[4],
      s_ws, l_ws, b_ws);

  fcos_topk_kernel<<<B_IMG, 256, 0, stream>>>(s_ws, l_ws, b_ws, tk_s, tk_l, tk_b);

  fcos_nms_kernel<<<B_IMG, 256, 0, stream>>>(tk_s, tk_l, tk_b, out);
}
